// U_GAT_2413771621184
// MI455X (gfx1250) — matrix operand order repaired and checked by reading, not compiled, not run on hardware
//
#include <hip/hip_runtime.h>

#define N_TRAIT 4
#define N_HEAD 8
#define D_MODEL 512
#define D_HEAD 64
#define BATCH 16384
#define MROWS (BATCH * N_TRAIT) /* 65536 */

#define BM 128
#define BN 128
#define BK 32
#define BKP 40 /* padded K stride in halves: 80B rows -> 16B aligned, bank-spread */

typedef __attribute__((ext_vector_type(16))) __bf16 v16bf;
typedef __attribute__((ext_vector_type(8)))  __bf16 v8bf;
typedef __attribute__((ext_vector_type(8)))  float  v8f;

union FragBF { v16bf v; v8bf h[2]; };

__device__ __forceinline__ unsigned short f2bf_bits(float x) {
  __bf16 h = (__bf16)x;
  return __builtin_bit_cast(unsigned short, h);
}
__device__ __forceinline__ float bf_bits2f(unsigned short u) {
  return (float)__builtin_bit_cast(__bf16, u);
}

// CDNA5 async global->LDS copy (ASYNCcnt-tracked); GV mode, 64-bit vaddr.
__device__ __forceinline__ void async_copy_b128(unsigned lds_off, const void* g) {
  asm volatile("global_load_async_to_lds_b128 %0, %1, off"
               :: "v"(lds_off), "v"((unsigned long long)(uintptr_t)g)
               : "memory");
}
__device__ __forceinline__ void wait_async0() {
  asm volatile("s_wait_asynccnt 0" ::: "memory");
}

// ---------------------------------------------------------------------------
// One-time: split W (fp32, row-major [n][k]) into bf16 hi/lo planes.
// ---------------------------------------------------------------------------
__global__ __launch_bounds__(256) void convert_w_kernel(
    const float* __restrict__ W, unsigned short* __restrict__ hi,
    unsigned short* __restrict__ lo) {
  const int i = blockIdx.x * 256 + threadIdx.x;
  const float x = W[i];
  const unsigned short hb = f2bf_bits(x);
  hi[i] = hb;
  lo[i] = f2bf_bits(x - bf_bits2f(hb));
}

// Build bf16 hi/lo WMMA A-fragments from 16 fp32 values held in registers.
// xs order matches A-frag half order: K = grp*8+e (e<8), K = 16+grp*8+(e-8).
__device__ __forceinline__ void build_frags(const float4 ar[4], FragBF& fh,
                                            FragBF& fl) {
  float xs[16];
  xs[0] = ar[0].x;  xs[1] = ar[0].y;  xs[2] = ar[0].z;  xs[3] = ar[0].w;
  xs[4] = ar[1].x;  xs[5] = ar[1].y;  xs[6] = ar[1].z;  xs[7] = ar[1].w;
  xs[8] = ar[2].x;  xs[9] = ar[2].y;  xs[10] = ar[2].z; xs[11] = ar[2].w;
  xs[12] = ar[3].x; xs[13] = ar[3].y; xs[14] = ar[3].z; xs[15] = ar[3].w;
#pragma unroll
  for (int e = 0; e < 16; ++e) {
    const __bf16 h = (__bf16)xs[e];
    fh.v[e] = h;
    fl.v[e] = (__bf16)(xs[e] - (float)h);
  }
}

// ---------------------------------------------------------------------------
// qk = feature @ W^T  via split-bf16 WMMA (hi*hi + hi*lo + lo*hi), f32 accum.
// Block: 256 thr = 8 waves; tile 128(M) x 128(N); K step 32.
//  - B (pre-split bf16 W planes): async global->LDS, double-buffered (40KB).
//  - A (fp32): per-lane direct global loads -> in-register bf16 hi/lo frags,
//    software-pipelined one k-step ahead. One barrier per k-step.
// ---------------------------------------------------------------------------
__global__ __launch_bounds__(256) void gemm_qk_kernel(
    const float* __restrict__ A, const unsigned short* __restrict__ Bhi,
    const unsigned short* __restrict__ Blo, float* __restrict__ C) {
  __shared__ __align__(16) unsigned short sB[2][2][BN * BKP];  // [stage][hi/lo]

  const int tid  = threadIdx.x;
  const int lane = tid & 31;
  const int wid  = tid >> 5;
  const int wm = (wid & 3) * 32;   // wave M offset within block tile
  const int wn = (wid >> 2) * 64;  // wave N offset within block tile
  const int blockM = blockIdx.y * BM;
  const int blockN = blockIdx.x * BN;
  const int grp   = lane >> 4;
  const int mlane = lane & 15;

  // B async-loader coords: each thread owns a 16-half chunk of the tile.
  const int bn = tid >> 1;
  const int bc = (tid & 1) * 16;
  const size_t bgbase = (size_t)(blockN + bn) * D_MODEL + bc;
  const unsigned ldsH = (unsigned)(uintptr_t)&sB[0][0][bn * BKP + bc];
  const unsigned ldsL = (unsigned)(uintptr_t)&sB[0][1][bn * BKP + bc];
  const unsigned stageStride = (unsigned)(2 * BN * BKP * sizeof(unsigned short));

  v8f acc[2][4];
  const v8f zero = {0.f, 0.f, 0.f, 0.f, 0.f, 0.f, 0.f, 0.f};
#pragma unroll
  for (int i = 0; i < 2; ++i)
#pragma unroll
    for (int j = 0; j < 4; ++j) acc[i][j] = zero;

  // Per-lane A row base pointers (K offset grp*8 baked in).
  const float* arow[2];
  arow[0] = A + (size_t)(blockM + wm + mlane) * D_MODEL + grp * 8;
  arow[1] = arow[0] + (size_t)16 * D_MODEL;

#define ISSUE_B(stage, k0)                                                    \
  do {                                                                        \
    const unsigned so = (unsigned)(stage) * stageStride;                      \
    async_copy_b128(ldsH + so,      Bhi + bgbase + (k0));                     \
    async_copy_b128(ldsH + so + 16, Bhi + bgbase + (k0) + 8);                 \
    async_copy_b128(ldsL + so,      Blo + bgbase + (k0));                     \
    async_copy_b128(ldsL + so + 16, Blo + bgbase + (k0) + 8);                 \
  } while (0)

#define LOAD_A(k0, dst)                                                       \
  do {                                                                        \
    _Pragma("unroll") for (int i_ = 0; i_ < 2; ++i_) {                        \
      const float* p_ = arow[i_] + (k0);                                      \
      dst[i_][0] = *(const float4*)(p_);                                      \
      dst[i_][1] = *(const float4*)(p_ + 4);                                  \
      dst[i_][2] = *(const float4*)(p_ + 16);                                 \
      dst[i_][3] = *(const float4*)(p_ + 20);                                 \
    }                                                                         \
  } while (0)

  // Prologue: stage 0 in flight.
  float4 areg[2][4];
  ISSUE_B(0, 0);
  LOAD_A(0, areg);

  for (int k = 0; k < D_MODEL / BK; ++k) {
    const int cur = k & 1;
    wait_async0();     // our stage-k async B copies have landed in LDS
    __syncthreads();   // everyone's copies landed; prev readers of buf[nxt] done

    float4 anext[2][4];
    if (k + 1 < D_MODEL / BK) {
      ISSUE_B(cur ^ 1, (k + 1) * BK);  // runs under this step's WMMAs
      LOAD_A((k + 1) * BK, anext);     // latency hidden behind compute
    }

    // A fragments for this step (in-register bf16 hi/lo split).
    FragBF ah[2], al[2];
    build_frags(areg[0], ah[0], al[0]);
    build_frags(areg[1], ah[1], al[1]);

    const unsigned short* sbh = &sB[cur][0][0];
    const unsigned short* sbl = &sB[cur][1][0];
#pragma unroll
    for (int j = 0; j < 4; ++j) {
      const int ncol = wn + j * 16 + mlane;
      const int kb = grp * 8;
      FragBF bh, bl;
      bh.h[0] = *(const v8bf*)&sbh[ncol * BKP + kb];
      bh.h[1] = *(const v8bf*)&sbh[ncol * BKP + kb + 16];
      bl.h[0] = *(const v8bf*)&sbl[ncol * BKP + kb];
      bl.h[1] = *(const v8bf*)&sbl[ncol * BKP + kb + 16];
#pragma unroll
      for (int i = 0; i < 2; ++i) {
        acc[i][j] = __builtin_amdgcn_wmma_f32_16x16x32_bf16(
            false, ah[i].v, false, bh.v, (short)0, acc[i][j], false, false);
        acc[i][j] = __builtin_amdgcn_wmma_f32_16x16x32_bf16(
            false, ah[i].v, false, bl.v, (short)0, acc[i][j], false, false);
        acc[i][j] = __builtin_amdgcn_wmma_f32_16x16x32_bf16(
            false, al[i].v, false, bh.v, (short)0, acc[i][j], false, false);
      }
    }

    if (k + 1 < D_MODEL / BK) {
#pragma unroll
      for (int i = 0; i < 2; ++i)
#pragma unroll
        for (int q = 0; q < 4; ++q) areg[i][q] = anext[i][q];
    }
  }
#undef ISSUE_B
#undef LOAD_A

  // Epilogue: C layout VGPR r -> M = r + (lane>=16 ? 8 : 0), N = lane%16.
  const int mofs = (lane >> 4) * 8;
  const int col0 = blockN + mlane;
#pragma unroll
  for (int i = 0; i < 2; ++i)
#pragma unroll
    for (int j = 0; j < 4; ++j)
#pragma unroll
      for (int r = 0; r < 8; ++r) {
        const size_t row = (size_t)blockM + wm + i * 16 + mofs + r;
        C[row * D_MODEL + col0 + wn + j * 16] = acc[i][j][r];
      }
}

// ---------------------------------------------------------------------------
// Fused attention tail: one wave32 per batch element, 4 waves per block.
//   sq/sk dots -> tanh scores -> softmax over k -> hid -> residual + tanh.
// Safe when residual == out (pure per-element read-then-write).
// ---------------------------------------------------------------------------
__global__ __launch_bounds__(128) void attn_kernel(
    const float* __restrict__ qk, const float* residual,
    const float* __restrict__ attn_w, float* out) {
  __shared__ float s_w[N_HEAD][2 * D_HEAD];                  // 4 KB
  __shared__ float s_qk[4][32][68];                          // padded rows
  __shared__ float s_attn[4][N_TRAIT][N_TRAIT][N_HEAD];      // [wv][q][k][h]

  const int tid  = threadIdx.x;
  const int lane = tid & 31;
  const int wv   = tid >> 5;
  const int b    = blockIdx.x * 4 + wv;
  const size_t base = (size_t)b * (N_TRAIT * D_MODEL);

  for (int i = tid; i < N_HEAD * 2 * D_HEAD; i += 128)
    (&s_w[0][0])[i] = attn_w[i];

  for (int i = lane; i < 512; i += 32) {  // 2048 floats, float4 coalesced
    const int flat = i * 4;
    const int l = flat >> 6;   // (n,h) row 0..31
    const int d = flat & 63;
    *(float4*)&s_qk[wv][l][d] = *(const float4*)&qk[base + flat];
  }
  __syncthreads();

  // phase 1: lane = n*8 + h
  const int n = lane >> 3;
  const int h = lane & 7;
  float sq = 0.f, sk = 0.f;
#pragma unroll 8
  for (int d = 0; d < D_HEAD; ++d) {
    const float v = s_qk[wv][lane][d];
    sq += v * s_w[h][d];
    sk += v * s_w[h][D_HEAD + d];
  }
  float sc[4];
  float m = -1e30f;
#pragma unroll
  for (int k = 0; k < 4; ++k) {
    const float skk = __shfl(sk, k * 8 + h, 32);  // sk[k,h] from lane k*8+h
    sc[k] = tanhf(sq + skk);
    m = fmaxf(m, sc[k]);
  }
  float ssum = 0.f;
#pragma unroll
  for (int k = 0; k < 4; ++k) { sc[k] = __expf(sc[k] - m); ssum += sc[k]; }
  const float inv = 1.f / ssum;
#pragma unroll
  for (int k = 0; k < 4; ++k) s_attn[wv][n][k][h] = sc[k] * inv;
  __syncthreads();

  // phase 2: out[b, q, h*64+d] = residual + tanh(sum_k attn[q,k,h]*qk[k,h,d])
#pragma unroll 4
  for (int t = 0; t < 16; ++t) {
    const int flat = (lane + t * 32) * 4;
    const int q  = flat >> 9;
    const int h2 = (flat >> 6) & 7;
    const int d  = flat & 63;
    const float w0 = s_attn[wv][q][0][h2];
    const float w1 = s_attn[wv][q][1][h2];
    const float w2 = s_attn[wv][q][2][h2];
    const float w3 = s_attn[wv][q][3][h2];
    const float4 res = *(const float4*)&residual[base + flat];
    const float rp[4] = {res.x, res.y, res.z, res.w};
    float op[4];
#pragma unroll
    for (int c = 0; c < 4; ++c) {
      const float v = w0 * s_qk[wv][0 * 8 + h2][d + c]
                    + w1 * s_qk[wv][1 * 8 + h2][d + c]
                    + w2 * s_qk[wv][2 * 8 + h2][d + c]
                    + w3 * s_qk[wv][3 * 8 + h2][d + c];
      op[c] = rp[c] + tanhf(v);
    }
    const float4 o = {op[0], op[1], op[2], op[3]};
    *(float4*)&out[base + flat] = o;
  }
}

// ---------------------------------------------------------------------------
extern "C" void kernel_launch(void* const* d_in, const int* in_sizes, int n_in,
                              void* d_out, int out_size, void* d_ws,
                              size_t ws_size, hipStream_t stream) {
  (void)in_sizes; (void)n_in; (void)out_size; (void)ws_size;
  const float* feature = (const float*)d_in[0];
  const float* W       = (const float*)d_in[1];
  const float* attn_w  = (const float*)d_in[2];
  float* out = (float*)d_out;

  char* ws = (char*)d_ws;
  unsigned short* Whi = (unsigned short*)ws;
  unsigned short* Wlo =
      (unsigned short*)(ws + (size_t)D_MODEL * D_MODEL * sizeof(unsigned short));
  float* qk =
      (float*)(ws + 2 * (size_t)D_MODEL * D_MODEL * sizeof(unsigned short));

  convert_w_kernel<<<(D_MODEL * D_MODEL) / 256, 256, 0, stream>>>(W, Whi, Wlo);

  const dim3 ggrid(D_MODEL / BN, MROWS / BM);  // (4, 512)

  // layer 0: feature -> qk -> d_out (feature_1)
  gemm_qk_kernel<<<ggrid, 256, 0, stream>>>(feature, Whi, Wlo, qk);
  attn_kernel<<<BATCH / 4, 128, 0, stream>>>(qk, feature, attn_w, out);

  // layer 1: d_out -> qk -> d_out (in-place residual is per-element safe)
  gemm_qk_kernel<<<ggrid, 256, 0, stream>>>(out, Whi, Wlo, qk);
  attn_kernel<<<BATCH / 4, 128, 0, stream>>>(qk, out, attn_w, out);
}
